// HeadVer4_39977555591274
// MI455X (gfx1250) — matrix operand order repaired and checked by reading, not compiled, not run on hardware
//
#include <hip/hip_runtime.h>
#include <hip/hip_bf16.h>

// ---------------------------------------------------------------------------
// Causal attention head (B=8, T=2048, C=1024, H=128), fp32 in/out.
// Pass 0: transpose+convert W{q,k,v} -> Wt[3][H][C] f16 (one-time, 768 KB).
// Pass 1: fused QKV projection: each A-fragment of x feeds 3 WMMAs (Q,K,V).
//         Q,K stored (B*T,H) f16 row-major; V stored transposed (B,H,T) f16.
// Pass 2: fused flash attention, one wave per 16-query tile, online softmax,
//         WMMA for QK^T and PV, LDS round-trip to re-layout P (C->A layout).
// ---------------------------------------------------------------------------

typedef _Float16 half_t;
typedef __attribute__((ext_vector_type(16))) _Float16 v16h;
typedef __attribute__((ext_vector_type(8)))  _Float16 h8;
typedef __attribute__((ext_vector_type(8)))  float    v8f;
typedef __attribute__((ext_vector_type(4)))  float    f4;

static constexpr int Bn = 8;
static constexpr int Tn = 2048;
static constexpr int Cn = 1024;
static constexpr int Hn = 128;
static constexpr int ATT_WAVES = 4;

__device__ __forceinline__ v16h pack16(h8 lo, h8 hi) {
  v16h r;
#pragma unroll
  for (int i = 0; i < 8; ++i) { r[i] = lo[i]; r[8 + i] = hi[i]; }
  return r;
}

// ---------------------------------------------------------------------------
// Pass 0: Wt[w][h][c] = (f16) W_w[c][h]   (w in {q,k,v})
// ---------------------------------------------------------------------------
__global__ __launch_bounds__(256) void wcvt_kernel(
    const float* __restrict__ Wq, const float* __restrict__ Wk,
    const float* __restrict__ Wv, half_t* __restrict__ Wt)
{
  int idx = blockIdx.x * 256 + threadIdx.x;        // 3*C*H total
  int c     = idx & (Cn - 1);
  int h     = (idx >> 10) & (Hn - 1);
  int which = idx >> 17;
  const float* __restrict__ W = (which == 0) ? Wq : (which == 1 ? Wk : Wv);
  Wt[idx] = (half_t)W[(size_t)c * Hn + h];
}

// ---------------------------------------------------------------------------
// Pass 1: fused projection. grid = B*T/16 blocks, 8 waves; wave w owns output
// cols [16w,16w+16) of all three matrices. One A-frag -> 3 WMMAs.
// ---------------------------------------------------------------------------
__global__ __launch_bounds__(256) void proj_kernel(
    const float*  __restrict__ x,
    const half_t* __restrict__ Wt,   // [3][H][C] f16
    half_t* __restrict__ Qh,         // (B*T, H) row-major f16
    half_t* __restrict__ Kh,         // (B*T, H) row-major f16
    half_t* __restrict__ Vt)         // (B, H, T) f16
{
  const int wave = threadIdx.x >> 5;
  const int lane = threadIdx.x & 31;
  const int hh   = lane >> 4;
  const int nl   = lane & 15;
  const int m0   = blockIdx.x * 16;   // row tile in flattened (B*T)
  const int n0   = wave * 16;         // col tile in H

  v8f accq = {}, acck = {}, accv = {};

  // A-frag row: m = m0+nl; element e <-> K = kk + 16*hh + e within each 32-chunk
  const float* xrow = x + (size_t)(m0 + nl) * Cn + 16 * hh;
  // B-frag rows (Wt): element e <-> K = kk + 16*hh + e, contiguous in C
  const half_t* wqr = Wt + ((size_t)(0 * Hn) + n0 + nl) * Cn + 16 * hh;
  const half_t* wkr = Wt + ((size_t)(1 * Hn) + n0 + nl) * Cn + 16 * hh;
  const half_t* wvr = Wt + ((size_t)(2 * Hn) + n0 + nl) * Cn + 16 * hh;

  for (int kk = 0; kk < Cn; kk += 32) {
    f4 xa0 = *(const f4*)(xrow + kk);
    f4 xa1 = *(const f4*)(xrow + kk + 4);
    f4 xb0 = *(const f4*)(xrow + kk + 8);
    f4 xb1 = *(const f4*)(xrow + kk + 12);
    v16h a;
#pragma unroll
    for (int i = 0; i < 4; ++i) {
      a[i]      = (half_t)xa0[i];
      a[4 + i]  = (half_t)xa1[i];
      a[8 + i]  = (half_t)xb0[i];
      a[12 + i] = (half_t)xb1[i];
    }
    v16h bq = pack16(*(const h8*)(wqr + kk), *(const h8*)(wqr + kk + 8));
    accq = __builtin_amdgcn_wmma_f32_16x16x32_f16(
        false, a, false, bq, (short)0, accq, false, false);
    v16h bk = pack16(*(const h8*)(wkr + kk), *(const h8*)(wkr + kk + 8));
    acck = __builtin_amdgcn_wmma_f32_16x16x32_f16(
        false, a, false, bk, (short)0, acck, false, false);
    v16h bv = pack16(*(const h8*)(wvr + kk), *(const h8*)(wvr + kk + 8));
    accv = __builtin_amdgcn_wmma_f32_16x16x32_f16(
        false, a, false, bv, (short)0, accv, false, false);
  }

  // ---- stores: C/D layout -> VGPR j holds row (j + 8*hh), col nl ----
#pragma unroll
  for (int j = 0; j < 8; ++j) {
    const int g = m0 + j + 8 * hh;                 // global row in B*T
    Qh[(size_t)g * Hn + n0 + nl] = (half_t)accq[j];
    Kh[(size_t)g * Hn + n0 + nl] = (half_t)acck[j];
    const int b = g >> 11;                         // / Tn
    const int t = g & (Tn - 1);
    Vt[((size_t)b * Hn + n0 + nl) * Tn + t] = (half_t)accv[j];
  }
}

// ---------------------------------------------------------------------------
// Pass 2: fused causal flash attention.
// grid = (T/(16*ATT_WAVES), B), block = 32*ATT_WAVES; each wave owns a
// 16-query tile and streams keys/values in 32-wide chunks.
// ---------------------------------------------------------------------------
__global__ __launch_bounds__(32 * ATT_WAVES) void attn_kernel(
    const half_t* __restrict__ Qh,
    const half_t* __restrict__ Kh,
    const half_t* __restrict__ Vt,
    float* __restrict__ out)
{
  __shared__ __align__(16) _Float16 Plds[ATT_WAVES][16 * 32];

  const int wave = threadIdx.x >> 5;
  const int lane = threadIdx.x & 31;
  const int hh   = lane >> 4;
  const int nl   = lane & 15;
  const int b    = blockIdx.y;
  const int q0   = (blockIdx.x * ATT_WAVES + wave) * 16;
  _Float16* __restrict__ P = &Plds[wave][0];

  const float scale = 0.03125f;  // C^-0.5 = 1/32

  // ---- resident Q fragments: 4 chunks of K=32 over head dim ----
  v16h qf[4];
  {
    const half_t* qrow = Qh + ((size_t)b * Tn + q0 + nl) * Hn + 16 * hh;
#pragma unroll
    for (int kc = 0; kc < 4; ++kc) {
      h8 lo = *(const h8*)(qrow + 32 * kc);
      h8 hi = *(const h8*)(qrow + 32 * kc + 8);
      qf[kc] = pack16(lo, hi);
    }
  }

  float mrow[8], lrow[8];
  v8f o[8];
#pragma unroll
  for (int j = 0; j < 8; ++j) { mrow[j] = -1.0e30f; lrow[j] = 0.0f; }
#pragma unroll
  for (int t = 0; t < 8; ++t) o[t] = (v8f){0.f,0.f,0.f,0.f,0.f,0.f,0.f,0.f};

  for (int s0 = 0; s0 < q0 + 16; s0 += 32) {
    // ---- S = Q K^T for two 16-key tiles (8 WMMAs) ----
    v8f sc0 = {}, sc1 = {};
    const half_t* k0row = Kh + ((size_t)b * Tn + s0 + nl) * Hn + 16 * hh;
    const half_t* k1row = k0row + (size_t)16 * Hn;
    if (s0 + 32 < q0 + 16) {  // prefetch next key chunk into cache hierarchy
      __builtin_prefetch(k0row + (size_t)32 * Hn, 0, 1);
      __builtin_prefetch(k1row + (size_t)32 * Hn, 0, 1);
    }
#pragma unroll
    for (int kc = 0; kc < 4; ++kc) {
      h8 a0 = *(const h8*)(k0row + 32 * kc);
      h8 a1 = *(const h8*)(k0row + 32 * kc + 8);
      v16h kf0 = pack16(a0, a1);
      sc0 = __builtin_amdgcn_wmma_f32_16x16x32_f16(
          false, qf[kc], false, kf0, (short)0, sc0, false, false);
      h8 b0 = *(const h8*)(k1row + 32 * kc);
      h8 b1 = *(const h8*)(k1row + 32 * kc + 8);
      v16h kf1 = pack16(b0, b1);
      sc1 = __builtin_amdgcn_wmma_f32_16x16x32_f16(
          false, qf[kc], false, kf1, (short)0, sc1, false, false);
    }

    // ---- online softmax update (row j+8*hh, 16 cols across lanes) ----
    float pr0[8], pr1[8];
#pragma unroll
    for (int j = 0; j < 8; ++j) {
      const int qg = q0 + j + 8 * hh;
      float v0 = sc0[j] * scale;
      float v1 = sc1[j] * scale;
      if (s0 + nl > qg)      v0 = -1.0e30f;   // causal mask
      if (s0 + 16 + nl > qg) v1 = -1.0e30f;
      float rm = fmaxf(v0, v1);
#pragma unroll
      for (int m = 1; m < 16; m <<= 1) rm = fmaxf(rm, __shfl_xor(rm, m, 32));
      const float mn    = fmaxf(mrow[j], rm);
      const float alpha = __expf(mrow[j] - mn);
      const float p0 = __expf(v0 - mn);
      const float p1 = __expf(v1 - mn);
      float rs = p0 + p1;
#pragma unroll
      for (int m = 1; m < 16; m <<= 1) rs += __shfl_xor(rs, m, 32);
      lrow[j] = lrow[j] * alpha + rs;
      mrow[j] = mn;
      pr0[j] = p0;
      pr1[j] = p1;
#pragma unroll
      for (int t = 0; t < 8; ++t) o[t][j] *= alpha;
    }

    // ---- P: C-layout -> LDS (row-major 16x32) -> A-layout fragment ----
#pragma unroll
    for (int j = 0; j < 8; ++j) {
      P[(j + 8 * hh) * 32 + nl]      = (half_t)pr0[j];
      P[(j + 8 * hh) * 32 + 16 + nl] = (half_t)pr1[j];
    }
    asm volatile("s_wait_dscnt 0" ::: "memory");  // per-wave LDS RAW fence
    v16h af;
    {
      h8 lo = *(const h8*)(P + nl * 32 + 16 * hh);
      h8 hi = *(const h8*)(P + nl * 32 + 16 * hh + 8);
      af = pack16(lo, hi);
    }

    // ---- O += P V  (8 WMMAs across head dim) ----
#pragma unroll
    for (int t = 0; t < 8; ++t) {
      const half_t* vrow =
          Vt + ((size_t)b * Hn + 16 * t + nl) * Tn + s0 + 16 * hh;
      h8 v0 = *(const h8*)(vrow);
      h8 v1 = *(const h8*)(vrow + 8);
      v16h vf = pack16(v0, v1);
      o[t] = __builtin_amdgcn_wmma_f32_16x16x32_f16(
          false, af, false, vf, (short)0, o[t], false, false);
    }
  }

  // ---- epilogue: O / l ----
#pragma unroll
  for (int j = 0; j < 8; ++j) {
    const float inv = 1.0f / lrow[j];
    const int qg = q0 + j + 8 * hh;
#pragma unroll
    for (int t = 0; t < 8; ++t)
      out[((size_t)b * Tn + qg) * Hn + 16 * t + nl] = o[t][j] * inv;
  }
}

// ---------------------------------------------------------------------------
extern "C" void kernel_launch(void* const* d_in, const int* in_sizes, int n_in,
                              void* d_out, int out_size, void* d_ws, size_t ws_size,
                              hipStream_t stream) {
  const float* x  = (const float*)d_in[0];
  const float* Wq = (const float*)d_in[1];
  const float* Wk = (const float*)d_in[2];
  const float* Wv = (const float*)d_in[3];

  half_t* Qh = (half_t*)d_ws;                       // 4 MB
  half_t* Kh = Qh + (size_t)Bn * Tn * Hn;           // 4 MB
  half_t* Vt = Kh + (size_t)Bn * Tn * Hn;           // 4 MB
  half_t* Wt = Vt + (size_t)Bn * Tn * Hn;           // 768 KB, [3][H][C]

  wcvt_kernel<<<(3 * Cn * Hn) / 256, 256, 0, stream>>>(Wq, Wk, Wv, Wt);

  proj_kernel<<<(Bn * Tn) / 16, 256, 0, stream>>>(x, Wt, Qh, Kh, Vt);

  attn_kernel<<<dim3(Tn / (16 * ATT_WAVES), Bn), 32 * ATT_WAVES, 0, stream>>>(
      Qh, Kh, Vt, (float*)d_out);
}
